// RITS_55490977464472
// MI455X (gfx1250) — matrix operand order repaired and checked by reading, not compiled, not run on hardware
//
#include <hip/hip_runtime.h>
#include <hip/hip_bf16.h>

// ---------------------------------------------------------------------------
// RITS forward for MI455X (gfx1250, wave32, WMMA).
//   * Precompute delta-only branches (gamma_h, gamma_x, alpha) as big parallel
//     bf16 WMMA GEMMs over all B*T rows (register-blocked 2Mx4N to cut L2
//     traffic 4x on the A operand).
//   * Serial scan: 3 WMMA GEMMs per step (x_h, z_h, gates[K=768,N=2048,
//     blocked 2Mx2N]) + elementwise LSTM update fused with next-step decay.
//   * bf16 operands, f32 accumulate (v_wmma_f32_16x16x32_bf16).
// ---------------------------------------------------------------------------

typedef __bf16 bf16;
typedef __attribute__((ext_vector_type(16))) __bf16 v16bf;
typedef __attribute__((ext_vector_type(8)))  __bf16 v8bf;
typedef __attribute__((ext_vector_type(8)))  float  v8f;

#define B_  512
#define T_  256
#define F_  128
#define H_  512
#define BT_ (B_ * T_)
#define KG_ 768           // gates K = F (c_c) + F (m) + H (hdec)
#define NG_ 2048          // gates N = 4H
#define ABUF_LD 768       // [c_c(128) | m(128) | hdec(512)]

// -------------------- WMMA fragment helpers (bf16, 16x16x32) ---------------
// A fragment (16x32, row-major source): lane L holds row M=(L&15);
//   K = {kh*8 .. kh*8+7} and {16+kh*8 .. 16+kh*8+7}, kh = L>>4.
__device__ __forceinline__ v16bf load_a_frag(const bf16* A, int lda, int tm,
                                             int kk, int lane) {
  int lo = lane & 15, kh = lane >> 4;
  const bf16* p = A + (size_t)(tm + lo) * lda + kk + kh * 8;
  v8bf a0 = *reinterpret_cast<const v8bf*>(p);
  v8bf a1 = *reinterpret_cast<const v8bf*>(p + 16);
  union { v16bf v; v8bf h[2]; } u;
  u.h[0] = a0; u.h[1] = a1;
  return u.v;
}

// B fragment (32x16): we compute A * W^T with W row-major [N x K]; lane L
// supplies column N=(L&15), K = {kh*8 .. kh*8+7} and {16+kh*8 .. 16+kh*8+7}.
__device__ __forceinline__ v16bf load_b_frag(const bf16* W, int ldw, int tn,
                                             int kk, int lane) {
  int lo = lane & 15, kh = lane >> 4;
  const bf16* p = W + (size_t)(tn + lo) * ldw + kk + kh * 8;
  v8bf b0 = *reinterpret_cast<const v8bf*>(p);
  v8bf b1 = *reinterpret_cast<const v8bf*>(p + 16);
  union { v16bf v; v8bf h[2]; } u;
  u.h[0] = b0; u.h[1] = b1;
  return u.v;
}

__device__ __forceinline__ v8f zero_v8f() {
  v8f z = {0.f, 0.f, 0.f, 0.f, 0.f, 0.f, 0.f, 0.f};
  return z;
}

// Single 16x16 tile (for the tiny latency-critical step GEMMs).
__device__ __forceinline__ v8f wmma_tile(const bf16* A, int lda,
                                         const bf16* W, int ldw,
                                         int tm, int tn, int K, int lane) {
  v8f acc = zero_v8f();
  for (int kk = 0; kk < K; kk += 32) {
    v16bf a = load_a_frag(A, lda, tm, kk, lane);
    v16bf b = load_b_frag(W, ldw, tn, kk, lane);
    acc = __builtin_amdgcn_wmma_f32_16x16x32_bf16(false, a, false, b,
                                                  (short)0, acc, false, false);
  }
  return acc;
}

// Register-blocked MI x NJ block of 16x16 tiles: A fragments reused across NJ
// columns, B fragments reused across MI rows (cuts L2 traffic MI/NJ-fold).
template <int MI, int NJ>
__device__ __forceinline__ void wmma_block(const bf16* A, int lda,
                                           const bf16* W, int ldw,
                                           int tm, int tn, int K, int lane,
                                           v8f acc[MI * NJ]) {
  for (int kk = 0; kk < K; kk += 32) {
    v16bf a[MI], b[NJ];
#pragma unroll
    for (int i = 0; i < MI; ++i) a[i] = load_a_frag(A, lda, tm + 16 * i, kk, lane);
#pragma unroll
    for (int j = 0; j < NJ; ++j) b[j] = load_b_frag(W, ldw, tn + 16 * j, kk, lane);
#pragma unroll
    for (int i = 0; i < MI; ++i)
#pragma unroll
      for (int j = 0; j < NJ; ++j)
        acc[i * NJ + j] = __builtin_amdgcn_wmma_f32_16x16x32_bf16(
            false, a[i], false, b[j], (short)0, acc[i * NJ + j], false, false);
  }
}

__device__ __forceinline__ float sigmoidf_(float x) {
  return 1.f / (1.f + __expf(-x));
}

// ----------------------------- small prep kernels ---------------------------
__global__ __launch_bounds__(256) void k_f32_to_bf16(const float* __restrict__ s,
                                                     bf16* __restrict__ d, int n) {
  int i = blockIdx.x * blockDim.x + threadIdx.x;
  if (i < n) d[i] = (bf16)s[i];
}

// Wg[n][k] = k<256 ? W_ih[n][k] : W_hh[n][k-256]   (2048 x 768)
__global__ __launch_bounds__(256) void k_build_wg(const float* __restrict__ Wih,
                                                  const float* __restrict__ Whh,
                                                  bf16* __restrict__ Wg) {
  int i = blockIdx.x * blockDim.x + threadIdx.x;
  if (i >= NG_ * KG_) return;
  int r = i / KG_, k = i - r * KG_;
  float v = (k < 256) ? Wih[r * 256 + k] : Whh[r * 512 + (k - 256)];
  Wg[i] = (bf16)v;
}

// bg = b_ih + b_hh ; c := 0 ; Abuf hdec region := 0
__global__ __launch_bounds__(256) void k_bias_init(const float* __restrict__ bih,
                                                   const float* __restrict__ bhh,
                                                   float* __restrict__ bg,
                                                   float* __restrict__ c,
                                                   bf16* __restrict__ Abuf) {
  int i = blockIdx.x * blockDim.x + threadIdx.x;
  if (i < NG_) bg[i] = bih[i] + bhh[i];
  if (i < B_ * H_) {
    c[i] = 0.f;
    int b = i >> 9, h = i & (H_ - 1);
    Abuf[(size_t)b * ABUF_LD + 256 + h] = (bf16)0.f;
  }
}

// alphaA[:, 128:256] = masks  (rows r = b*T+t)
__global__ __launch_bounds__(256) void k_mask_fill(const int* __restrict__ masks,
                                                   bf16* __restrict__ alphaA) {
  int i = blockIdx.x * blockDim.x + threadIdx.x;
  if (i >= BT_ * F_) return;
  int r = i >> 7, f = i & (F_ - 1);
  alphaA[(size_t)r * 256 + 128 + f] = (bf16)(float)masks[i];
}

// --------------------------- precompute GEMMs (2Mx4N) -----------------------
// out[m][n] = exp(-relu(A.W^T + bias))  -> bf16, row stride ldo
__global__ __launch_bounds__(256) void k_gemm_expnegrelu(
    const bf16* __restrict__ A, int lda, const bf16* __restrict__ W, int ldw,
    const float* __restrict__ bias, bf16* __restrict__ out, int ldo,
    int N, int K) {
  int lane = threadIdx.x & 31, wave = threadIdx.x >> 5;
  int unit = blockIdx.x * 8 + wave;
  int unitsN = N >> 6;                 // N / 64
  int tm = (unit / unitsN) << 5;       // 32 rows
  int tn = (unit % unitsN) << 6;       // 64 cols
  v8f acc[8];
#pragma unroll
  for (int q = 0; q < 8; ++q) acc[q] = zero_v8f();
  wmma_block<2, 4>(A, lda, W, ldw, tm, tn, K, lane, acc);
  int lo = lane & 15, hi = lane >> 4;
#pragma unroll
  for (int j = 0; j < 4; ++j) {
    int n = tn + j * 16 + lo;
    float bn = bias[n];
#pragma unroll
    for (int i = 0; i < 2; ++i) {
#pragma unroll
      for (int r = 0; r < 8; ++r) {
        int m = tm + i * 16 + hi * 8 + r;
        float v = acc[i * 4 + j][r] + bn;
        v = v > 0.f ? v : 0.f;
        out[(size_t)m * ldo + n] = (bf16)__expf(-v);
      }
    }
  }
}

// alpha[m][n] = alphaA.Wwc^T + b_wc   (M=BT, N=128, K=256) -- 2Mx4N blocked
__global__ __launch_bounds__(256) void k_gemm_alpha(
    const bf16* __restrict__ A, const bf16* __restrict__ W,
    const float* __restrict__ bias, bf16* __restrict__ out) {
  int lane = threadIdx.x & 31, wave = threadIdx.x >> 5;
  int unit = blockIdx.x * 8 + wave;
  int tm = (unit >> 1) << 5;           // unitsN = 128/64 = 2
  int tn = (unit & 1) << 6;
  v8f acc[8];
#pragma unroll
  for (int q = 0; q < 8; ++q) acc[q] = zero_v8f();
  wmma_block<2, 4>(A, 256, W, 256, tm, tn, 256, lane, acc);
  int lo = lane & 15, hi = lane >> 4;
#pragma unroll
  for (int j = 0; j < 4; ++j) {
    int n = tn + j * 16 + lo;
    float bn = bias[n];
#pragma unroll
    for (int i = 0; i < 2; ++i) {
#pragma unroll
      for (int r = 0; r < 8; ++r) {
        int m = tm + i * 16 + hi * 8 + r;
        out[(size_t)m * F_ + n] = (bf16)(acc[i * 4 + j][r] + bn);
      }
    }
  }
}

// ------------------------------ per-step GEMMs ------------------------------
// x_h = hdec.Whr^T + b_hr ; x_c = m?x:x_h   (M=512, N=128, K=512; 1 tile/wave)
__global__ __launch_bounds__(256) void k_step_xh(
    const bf16* __restrict__ Abuf, const bf16* __restrict__ Whr,
    const float* __restrict__ bhr, const float* __restrict__ values,
    const int* __restrict__ masks, int t,
    float* __restrict__ xh, bf16* __restrict__ xc) {
  int lane = threadIdx.x & 31, wave = threadIdx.x >> 5;
  int tile = blockIdx.x * 8 + wave;
  int tm = (tile >> 3) << 4;          // tilesN = 8
  int tn = (tile & 7) << 4;
  v8f acc = wmma_tile(Abuf + 256, ABUF_LD, Whr, H_, tm, tn, H_, lane);
  int lo = lane & 15, hi = lane >> 4;
  int n = tn + lo;
  float bn = bhr[n];
#pragma unroll
  for (int r = 0; r < 8; ++r) {
    int b = tm + hi * 8 + r;
    float v = acc[r] + bn;
    xh[b * F_ + n] = v;
    int idx = (b * T_ + t) * F_ + n;
    float xv = values[idx];
    int mm = masks[idx];
    xc[b * F_ + n] = (bf16)(mm ? xv : v);
  }
}

// z_h = x_c.Wfr^T + b_fr ; c_h = a*z+(1-a)*x_h ; c_c = m?x:c_h -> out, Abuf
__global__ __launch_bounds__(256) void k_step_zh(
    const bf16* __restrict__ xc, const bf16* __restrict__ Wfr,
    const float* __restrict__ bfr, const float* __restrict__ values,
    const int* __restrict__ masks, const bf16* __restrict__ alpha,
    const float* __restrict__ xh, int t,
    float* __restrict__ out, bf16* __restrict__ Abuf) {
  int lane = threadIdx.x & 31, wave = threadIdx.x >> 5;
  int tile = blockIdx.x * 8 + wave;
  int tm = (tile >> 3) << 4;          // tilesN = 8
  int tn = (tile & 7) << 4;
  v8f acc = wmma_tile(xc, F_, Wfr, F_, tm, tn, F_, lane);
  int lo = lane & 15, hi = lane >> 4;
  int n = tn + lo;
  float bn = bfr[n];
#pragma unroll
  for (int r = 0; r < 8; ++r) {
    int b = tm + hi * 8 + r;
    float z = acc[r] + bn;
    float a = (float)alpha[((size_t)b * T_ + t) * F_ + n];
    float xhv = xh[b * F_ + n];
    float ch = a * z + (1.f - a) * xhv;
    int idx = (b * T_ + t) * F_ + n;
    float xv = values[idx];
    int mm = masks[idx];
    float cc = mm ? xv : ch;
    out[idx] = cc;
    Abuf[(size_t)b * ABUF_LD + n] = (bf16)cc;
    Abuf[(size_t)b * ABUF_LD + 128 + n] = (bf16)(float)mm;
  }
}

// gates = Abuf.Wg^T + bg   (M=512, N=2048, K=768) -- 2Mx2N blocked
__global__ __launch_bounds__(256) void k_step_gates(
    const bf16* __restrict__ Abuf, const bf16* __restrict__ Wg,
    const float* __restrict__ bg, float* __restrict__ gates) {
  int lane = threadIdx.x & 31, wave = threadIdx.x >> 5;
  int unit = blockIdx.x * 8 + wave;
  int tm = (unit >> 6) << 5;           // unitsN = 2048/32 = 64
  int tn = (unit & 63) << 5;
  v8f acc[4];
#pragma unroll
  for (int q = 0; q < 4; ++q) acc[q] = zero_v8f();
  wmma_block<2, 2>(Abuf, ABUF_LD, Wg, KG_, tm, tn, KG_, lane, acc);
  int lo = lane & 15, hi = lane >> 4;
#pragma unroll
  for (int j = 0; j < 2; ++j) {
    int n = tn + j * 16 + lo;
    float bn = bg[n];
#pragma unroll
    for (int i = 0; i < 2; ++i) {
#pragma unroll
      for (int r = 0; r < 8; ++r) {
        int b = tm + i * 16 + hi * 8 + r;
        gates[(size_t)b * NG_ + n] = acc[i * 2 + j][r] + bn;
      }
    }
  }
}

// LSTM update; also pre-decays h for step t+1: Abuf.hdec = h_new * gamma_h(t+1)
__global__ __launch_bounds__(256) void k_step_update(
    const float* __restrict__ gates, float* __restrict__ c,
    const bf16* __restrict__ gammaH, bf16* __restrict__ Abuf, int t) {
  int i = blockIdx.x * blockDim.x + threadIdx.x;
  if (i >= B_ * H_) return;
  int b = i >> 9, h = i & (H_ - 1);
  const float* g = gates + (size_t)b * NG_;
  float ig = g[h], fg = g[512 + h], gg = g[1024 + h], og = g[1536 + h];
  float cn = sigmoidf_(fg) * c[i] + sigmoidf_(ig) * tanhf(gg);
  float hn = sigmoidf_(og) * tanhf(cn);
  c[i] = cn;
  if (t + 1 < T_) {
    float gH = (float)gammaH[((size_t)b * T_ + (t + 1)) * H_ + h];
    Abuf[(size_t)b * ABUF_LD + 256 + h] = (bf16)(hn * gH);
  }
}

// ------------------------------- launcher -----------------------------------
extern "C" void kernel_launch(void* const* d_in, const int* in_sizes, int n_in,
                              void* d_out, int out_size, void* d_ws, size_t ws_size,
                              hipStream_t stream) {
  const float* values = (const float*)d_in[0];
  const int*   masks  = (const int*)d_in[1];
  const float* deltas = (const float*)d_in[2];
  const float* W_dh = (const float*)d_in[3];  const float* b_dh = (const float*)d_in[4];
  const float* W_dx = (const float*)d_in[5];  const float* b_dx = (const float*)d_in[6];
  const float* W_hr = (const float*)d_in[7];  const float* b_hr = (const float*)d_in[8];
  const float* W_fr = (const float*)d_in[9];  const float* b_fr = (const float*)d_in[10];
  const float* W_wc = (const float*)d_in[11]; const float* b_wc = (const float*)d_in[12];
  const float* W_ih = (const float*)d_in[13]; const float* W_hh = (const float*)d_in[14];
  const float* b_ih = (const float*)d_in[15]; const float* b_hh = (const float*)d_in[16];
  float* out = (float*)d_out;

  // workspace carve (256B aligned)
  char* p = (char*)d_ws;
  auto carve = [&](size_t bytes) -> char* {
    char* r = p;
    p += (bytes + 255) & ~(size_t)255;
    return r;
  };
  bf16* deltas_bf = (bf16*)carve((size_t)BT_ * F_ * 2);
  bf16* gammaH    = (bf16*)carve((size_t)BT_ * H_ * 2);
  bf16* alphaA    = (bf16*)carve((size_t)BT_ * 256 * 2);   // [gamma_x | m]
  bf16* alpha     = (bf16*)carve((size_t)BT_ * F_ * 2);
  bf16* Wdh = (bf16*)carve((size_t)H_ * F_ * 2);
  bf16* Wdx = (bf16*)carve((size_t)F_ * F_ * 2);
  bf16* Wwc = (bf16*)carve((size_t)F_ * 2 * F_ * 2);
  bf16* Whr = (bf16*)carve((size_t)F_ * H_ * 2);
  bf16* Wfr = (bf16*)carve((size_t)F_ * F_ * 2);
  bf16* Wg  = (bf16*)carve((size_t)NG_ * KG_ * 2);
  float* bg = (float*)carve((size_t)NG_ * 4);
  bf16* Abuf = (bf16*)carve((size_t)B_ * ABUF_LD * 2);
  float* xh  = (float*)carve((size_t)B_ * F_ * 4);
  bf16* xc   = (bf16*)carve((size_t)B_ * F_ * 2);
  float* gates = (float*)carve((size_t)B_ * NG_ * 4);
  float* cst   = (float*)carve((size_t)B_ * H_ * 4);

  // ---- prep: bf16 conversions & packing ----
  int n;
  n = BT_ * F_;      k_f32_to_bf16<<<(n + 255) / 256, 256, 0, stream>>>(deltas, deltas_bf, n);
  n = H_ * F_;       k_f32_to_bf16<<<(n + 255) / 256, 256, 0, stream>>>(W_dh, Wdh, n);
  n = F_ * F_;       k_f32_to_bf16<<<(n + 255) / 256, 256, 0, stream>>>(W_dx, Wdx, n);
  n = F_ * 2 * F_;   k_f32_to_bf16<<<(n + 255) / 256, 256, 0, stream>>>(W_wc, Wwc, n);
  n = F_ * H_;       k_f32_to_bf16<<<(n + 255) / 256, 256, 0, stream>>>(W_hr, Whr, n);
  n = F_ * F_;       k_f32_to_bf16<<<(n + 255) / 256, 256, 0, stream>>>(W_fr, Wfr, n);
  n = NG_ * KG_;     k_build_wg<<<(n + 255) / 256, 256, 0, stream>>>(W_ih, W_hh, Wg);
  k_bias_init<<<(B_ * H_ + 255) / 256, 256, 0, stream>>>(b_ih, b_hh, bg, cst, Abuf);
  n = BT_ * F_;      k_mask_fill<<<(n + 255) / 256, 256, 0, stream>>>(masks, alphaA);

  // ---- precompute parallel GEMMs over all B*T rows (2Mx4N blocked) ----
  // gamma_x -> alphaA[:, 0:128]  (M=BT, N=128, K=128): 4096*2 units -> 1024 blk
  k_gemm_expnegrelu<<<1024, 256, 0, stream>>>(deltas_bf, F_, Wdx, F_,
                                              b_dx, alphaA, 256, F_, F_);
  // gamma_h                      (M=BT, N=512, K=128): 4096*8 units -> 4096 blk
  k_gemm_expnegrelu<<<4096, 256, 0, stream>>>(deltas_bf, F_, Wdh, F_,
                                              b_dh, gammaH, H_, H_, F_);
  // alpha                        (M=BT, N=128, K=256): 8192 units -> 1024 blk
  k_gemm_alpha<<<1024, 256, 0, stream>>>(alphaA, Wwc, b_wc, alpha);

  // ---- sequential scan ----
  for (int t = 0; t < T_; ++t) {
    k_step_xh   <<<32,  256, 0, stream>>>(Abuf, Whr, b_hr, values, masks, t, xh, xc);
    k_step_zh   <<<32,  256, 0, stream>>>(xc, Wfr, b_fr, values, masks, alpha, xh, t, out, Abuf);
    k_step_gates<<<128, 256, 0, stream>>>(Abuf, Wg, bg, gates);   // 1024 units
    k_step_update<<<(B_ * H_ + 255) / 256, 256, 0, stream>>>(gates, cst, gammaH, Abuf, t);
  }
}
